// MultiHeadAttention_8005819040257
// MI455X (gfx1250) — compile-verified, matrix operand order checked by reading, not run on hardware
//
#include <hip/hip_runtime.h>
#include <hip/hip_bf16.h>
#include <stdint.h>

// ---------------------------------------------------------------------------
// MI455X (gfx1250) causal multi-head attention, bf16 WMMA with f32 accumulate.
// B=4, T=2048, C=1024, H=16, D=64.
// ---------------------------------------------------------------------------

typedef __attribute__((ext_vector_type(8)))  __bf16 bf16x8;
typedef __attribute__((ext_vector_type(16))) __bf16 bf16x16;
typedef __attribute__((ext_vector_type(8)))  float  f32x8;
typedef __attribute__((ext_vector_type(4)))  unsigned int u32x4;
typedef __attribute__((ext_vector_type(8)))  int i32x8;
typedef __attribute__((ext_vector_type(4)))  int i32x4;

#define WMMA_BF16(A, B, C) \
  __builtin_amdgcn_wmma_f32_16x16x32_bf16(false, (A), false, (B), (short)0, (C), false, false)

#if defined(__has_builtin)
#if __has_builtin(__builtin_amdgcn_tensor_load_to_lds)
#define HAVE_TDM 1
#endif
#endif

// Assemble a 16x32 A-frag / 32x16 B-frag (bf16) from two contiguous 16-byte
// chunks per lane (ISA 7.12.2: VGPR0-3 = K base..base+7, VGPR4-7 = base+16..+23).
__device__ __forceinline__ bf16x16 make_frag(const __bf16* p0, const __bf16* p1) {
  bf16x8 lo = *(const bf16x8*)p0;
  bf16x8 hi = *(const bf16x8*)p1;
  bf16x16 r;
#pragma unroll
  for (int i = 0; i < 8; ++i) { r[i] = lo[i]; r[i + 8] = hi[i]; }
  return r;
}

#ifdef HAVE_TDM
// TDM: DMA a [rows=64, cols=32] bf16 tile of a row-major [8192,1024] tensor
// into LDS at byte offset lds_off. D# layout per CDNA5 ISA.
__device__ __forceinline__ void tdm_load_tile_64x32(const __bf16* gptr, unsigned lds_off) {
  uint64_t ga = (uint64_t)(uintptr_t)gptr;
  u32x4 g0;
  g0[0] = 1u;                                   // count=1, user mode
  g0[1] = lds_off;                              // lds_addr (bytes)
  g0[2] = (unsigned)(ga & 0xffffffffu);         // global_addr[31:0]
  g0[3] = (unsigned)((ga >> 32) & 0x01ffffffu)  // global_addr[56:32]
        | (2u << 30);                           // type=2 ("image")
  i32x8 g1;
  g1[0] = (1 << 16);                 // data_size=1 (2 bytes); no mask/flags
  g1[1] = (int)(1024u << 16);        // tensor_dim0[15:0]=1024 at bits[63:48]
  g1[2] = (int)(8192u << 16);        // tensor_dim1[15:0]=8192 at bits[111:80] low half
  g1[3] = (int)(32u << 16);          // tile_dim0=32 at bits[127:112]
  g1[4] = 64;                        // tile_dim1=64 at bits[143:128]; tile_dim2=0
  g1[5] = 1024;                      // tensor_dim0_stride[31:0]=1024
  g1[6] = 0;                         // stride0 hi, stride1 lo (unused, 2D tile)
  g1[7] = 0;
  i32x4 gz = {0, 0, 0, 0};
#if __clang_major__ >= 23
  i32x8 gz8 = {0, 0, 0, 0, 0, 0, 0, 0};
  __builtin_amdgcn_tensor_load_to_lds(g0, g1, gz, gz, gz8, 0);
#else
  __builtin_amdgcn_tensor_load_to_lds(g0, g1, gz, gz, 0);
#endif
}
#endif

// ---------------------------------------------------------------------------
// fp32 -> bf16 cast
// ---------------------------------------------------------------------------
__global__ void cast_f32_bf16(const float* __restrict__ in, __bf16* __restrict__ out, int n) {
  int i = blockIdx.x * blockDim.x + threadIdx.x;
  if (i < n) out[i] = (__bf16)in[i];
}

// ---------------------------------------------------------------------------
// GEMM: Out[M=8192, N=1024] = X[M,1024] * W^T + bias  (W is [N,1024])
// Block: 256 threads = 8 waves; block tile 64(M) x 256(N); wave tile 64x32.
// X tile (64x32 bf16) staged in LDS via TDM (double-buffered); W frags direct.
// Per k-step per wave: 8 WMMAs : 4 global b128 (B) + 8 ds b128 (A).
// MODE 0: bf16 out, row-major [M,1024]           (Q, K)
// MODE 1: bf16 out, transposed Vt[B,H,D,T]       (V)
// MODE 2: f32  out, row-major [M,1024]           (output projection)
// ---------------------------------------------------------------------------
template <int MODE>
__global__ __launch_bounds__(256) void gemm_xwt(const __bf16* __restrict__ X,
                                                const __bf16* __restrict__ W,
                                                const float*  __restrict__ bias,
                                                void* __restrict__ outp) {
  __shared__ __bf16 lds[2][64 * 32];

  const int lane = threadIdx.x & 31;
  const int wv   = threadIdx.x >> 5;     // 0..7
  const int m0   = blockIdx.x * 64;
  const int n0   = blockIdx.y * 256;
  const int wn   = wv * 32;              // wave's 32 N-columns
  const int lh   = lane >> 4;            // half-wave id
  const int ll   = lane & 15;

  f32x8 acc[4][2];
#pragma unroll
  for (int nb = 0; nb < 2; ++nb) {
    float bv = bias[n0 + wn + nb * 16 + ll];
#pragma unroll
    for (int mb = 0; mb < 4; ++mb)
#pragma unroll
      for (int r = 0; r < 8; ++r) acc[mb][nb][r] = bv;
  }

  const __bf16* xrow = &X[m0 * 1024];

#ifdef HAVE_TDM
  const unsigned ldsoff0 = (unsigned)(uintptr_t)&lds[0][0];
  const unsigned ldsoff1 = (unsigned)(uintptr_t)&lds[1][0];
  if (threadIdx.x < 32) tdm_load_tile_64x32(xrow, ldsoff0);   // preload tile 0
#else
  const int ldr = threadIdx.x >> 2;        // 0..63 row of X tile
  const int ldc = (threadIdx.x & 3) * 8;   // 8-elt chunk
#endif

  for (int k0 = 0; k0 < 1024; k0 += 32) {
    const int cur = (k0 >> 5) & 1;
#ifdef HAVE_TDM
    if (threadIdx.x < 32) {
      if (k0 + 32 < 1024) {
        tdm_load_tile_64x32(xrow + (k0 + 32), cur ? ldsoff0 : ldsoff1);
        __builtin_amdgcn_s_wait_tensorcnt(1);   // tile `cur` done, next in flight
      } else {
        __builtin_amdgcn_s_wait_tensorcnt(0);   // last tile: drain
      }
    }
    __syncthreads();                       // lds[cur] visible to all waves
#else
    __syncthreads();
    *(bf16x8*)&lds[cur][ldr * 32 + ldc] = *(const bf16x8*)&xrow[ldr * 1024 + k0 + ldc];
    __syncthreads();
#endif

    if (k0 + 32 < 1024)
      __builtin_prefetch(&W[(n0 + wn + ll) * 1024 + k0 + 32], 0, 0);

    bf16x16 bfr[2], afr[4];
#pragma unroll
    for (int nb = 0; nb < 2; ++nb) {
      const __bf16* wp = &W[(n0 + wn + nb * 16 + ll) * 1024 + k0 + lh * 8];
      bfr[nb] = make_frag(wp, wp + 16);
    }
#pragma unroll
    for (int mb = 0; mb < 4; ++mb) {
      const __bf16* ap = &lds[cur][(mb * 16 + ll) * 32 + lh * 8];
      afr[mb] = make_frag(ap, ap + 16);
    }
#pragma unroll
    for (int mb = 0; mb < 4; ++mb)
#pragma unroll
      for (int nb = 0; nb < 2; ++nb)
        acc[mb][nb] = WMMA_BF16(afr[mb], bfr[nb], acc[mb][nb]);

    __syncthreads();                       // done reading lds[cur]
  }

#pragma unroll
  for (int mb = 0; mb < 4; ++mb)
#pragma unroll
    for (int nb = 0; nb < 2; ++nb) {
      const int ncol = n0 + wn + nb * 16 + ll;
      if (MODE == 0) {
        __bf16* o = (__bf16*)outp;
#pragma unroll
        for (int r = 0; r < 8; ++r) {
          int m = m0 + mb * 16 + r + 8 * lh;
          o[m * 1024 + ncol] = (__bf16)acc[mb][nb][r];
        }
      } else if (MODE == 1) {
        // Vt[(b*1024 + ncol)*2048 + t] ; lane holds 8 consecutive t's -> one 16B store
        __bf16* o = (__bf16*)outp;
        int mstart = m0 + mb * 16 + 8 * lh;
        int b = mstart >> 11;
        int t = mstart & 2047;
        bf16x8 pk;
#pragma unroll
        for (int r = 0; r < 8; ++r) pk[r] = (__bf16)acc[mb][nb][r];
        *(bf16x8*)&o[(b * 1024 + ncol) * 2048 + t] = pk;
      } else {
        float* o = (float*)outp;
#pragma unroll
        for (int r = 0; r < 8; ++r) {
          int m = m0 + mb * 16 + r + 8 * lh;
          o[m * 1024 + ncol] = acc[mb][nb][r];
        }
      }
    }
}

// ---------------------------------------------------------------------------
// Flash attention: one wave per (b, h, 16-query tile). Key tiles of 64.
// Q,K: bf16 [B*T, 1024] (head slice at col h*64). Vt: bf16 [B,H,D,T].
// Online softmax in f32; P staged via LDS to re-layout C-frag -> A-frag.
// ---------------------------------------------------------------------------
__global__ __launch_bounds__(128) void attn_fwd(const __bf16* __restrict__ Q,
                                                const __bf16* __restrict__ K,
                                                const __bf16* __restrict__ Vt,
                                                __bf16* __restrict__ O) {
  __shared__ __bf16 plds[4 * 16 * 64];

  const int lane = threadIdx.x & 31;
  const int wv   = threadIdx.x >> 5;
  const int gw   = blockIdx.x * 4 + wv;     // 0..8191
  const int qt   = gw & 127;                // T/16
  const int h    = (gw >> 7) & 15;
  const int b    = gw >> 11;
  const int q0   = qt * 16;
  const int lh   = lane >> 4;
  const int ll   = lane & 15;
  __bf16* pl = &plds[wv * 1024];

  const __bf16* Qh = Q + (b * 2048) * 1024 + h * 64;
  const __bf16* Kh = K + (b * 2048) * 1024 + h * 64;
  const __bf16* Vh = Vt + (b * 1024 + h * 64) * 2048;

  bf16x16 qa[2];
#pragma unroll
  for (int kc = 0; kc < 2; ++kc) {
    const __bf16* qp = &Qh[(q0 + ll) * 1024 + kc * 32 + lh * 8];
    qa[kc] = make_frag(qp, qp + 16);
  }

  f32x8 o[4];
#pragma unroll
  for (int ns = 0; ns < 4; ++ns)
#pragma unroll
    for (int r = 0; r < 8; ++r) o[ns][r] = 0.0f;

  float Mx[8], L[8];
#pragma unroll
  for (int r = 0; r < 8; ++r) { Mx[r] = -3.0e38f; L[r] = 0.0f; }

  const float scale = 0.125f;  // 1/sqrt(64)

  for (int kt = 0; kt <= q0 + 15; kt += 64) {
    // S = Q K^T for 16x64 tile
    f32x8 s[4];
#pragma unroll
    for (int ns = 0; ns < 4; ++ns) {
      f32x8 z;
#pragma unroll
      for (int r = 0; r < 8; ++r) z[r] = 0.0f;
#pragma unroll
      for (int kc = 0; kc < 2; ++kc) {
        const __bf16* kp = &Kh[(kt + ns * 16 + ll) * 1024 + kc * 32 + lh * 8];
        bf16x16 kb = make_frag(kp, kp + 16);
        z = WMMA_BF16(qa[kc], kb, z);
      }
      s[ns] = z;
    }

    // scale + causal mask + per-row max (row m = r + 8*lh lives in VGPR r across a half-wave)
    float rm[8];
#pragma unroll
    for (int r = 0; r < 8; ++r) {
      const int qi = q0 + r + 8 * lh;
      float mx = -3.0e38f;
#pragma unroll
      for (int ns = 0; ns < 4; ++ns) {
        const int ki = kt + ns * 16 + ll;
        float v = s[ns][r] * scale;
        v = (ki <= qi) ? v : -3.0e38f;
        s[ns][r] = v;
        mx = fmaxf(mx, v);
      }
      rm[r] = mx;
    }
#pragma unroll
    for (int d = 1; d < 16; d <<= 1)
#pragma unroll
      for (int r = 0; r < 8; ++r) rm[r] = fmaxf(rm[r], __shfl_xor(rm[r], d, 32));

    float alpha[8];
#pragma unroll
    for (int r = 0; r < 8; ++r) {
      float mn = fmaxf(Mx[r], rm[r]);
      alpha[r] = __expf(Mx[r] - mn);
      Mx[r] = mn;
    }

    // P = exp(S - M); write C-layout tile to LDS for A-layout reload
    float rs[8];
#pragma unroll
    for (int r = 0; r < 8; ++r) rs[r] = 0.0f;
#pragma unroll
    for (int ns = 0; ns < 4; ++ns)
#pragma unroll
      for (int r = 0; r < 8; ++r) {
        float p = __expf(s[ns][r] - Mx[r]);
        rs[r] += p;
        pl[(r + 8 * lh) * 64 + ns * 16 + ll] = (__bf16)p;
      }
#pragma unroll
    for (int d = 1; d < 16; d <<= 1)
#pragma unroll
      for (int r = 0; r < 8; ++r) rs[r] += __shfl_xor(rs[r], d, 32);
#pragma unroll
    for (int r = 0; r < 8; ++r) L[r] = L[r] * alpha[r] + rs[r];
#pragma unroll
    for (int ns = 0; ns < 4; ++ns)
#pragma unroll
      for (int r = 0; r < 8; ++r) o[ns][r] *= alpha[r];

    // LDS ops from this wave are in-order; fence vs compiler + wait store completion
    asm volatile("s_wait_dscnt 0" ::: "memory");

    bf16x16 pa[2];
#pragma unroll
    for (int kc = 0; kc < 2; ++kc) {
      const __bf16* pp = &pl[ll * 64 + kc * 32 + lh * 8];
      pa[kc] = make_frag(pp, pp + 16);
    }

    // O += P * V  (B-operand from pre-transposed Vt: contiguous along keys)
#pragma unroll
    for (int ns = 0; ns < 4; ++ns) {
#pragma unroll
      for (int kc = 0; kc < 2; ++kc) {
        const __bf16* vp = &Vh[(ns * 16 + ll) * 2048 + kt + kc * 32 + lh * 8];
        bf16x16 vb = make_frag(vp, vp + 16);
        o[ns] = WMMA_BF16(pa[kc], vb, o[ns]);
      }
    }
  }

  // normalize and store bf16 attention output in [B*T, 1024] layout
  float inv[8];
#pragma unroll
  for (int r = 0; r < 8; ++r) inv[r] = 1.0f / L[r];
#pragma unroll
  for (int ns = 0; ns < 4; ++ns)
#pragma unroll
    for (int r = 0; r < 8; ++r) {
      int row = b * 2048 + q0 + r + 8 * lh;
      O[row * 1024 + h * 64 + ns * 16 + ll] = (__bf16)(o[ns][r] * inv[r]);
    }
}

// ---------------------------------------------------------------------------
// Host-side launch
// ---------------------------------------------------------------------------
extern "C" void kernel_launch(void* const* d_in, const int* in_sizes, int n_in,
                              void* d_out, int out_size, void* d_ws, size_t ws_size,
                              hipStream_t stream) {
  const float* x  = (const float*)d_in[0];
  const float* Wq = (const float*)d_in[1];
  const float* bq = (const float*)d_in[2];
  const float* Wk = (const float*)d_in[3];
  const float* bk = (const float*)d_in[4];
  const float* Wv = (const float*)d_in[5];
  const float* bv = (const float*)d_in[6];
  const float* Wo = (const float*)d_in[7];
  const float* bo = (const float*)d_in[8];

  char* ws = (char*)d_ws;
  const size_t MB = 1u << 20;
  __bf16* Xb  = (__bf16*)(ws + 0 * MB);    // 16 MB  [8192,1024]
  __bf16* Wqb = (__bf16*)(ws + 16 * MB);   //  2 MB
  __bf16* Wkb = (__bf16*)(ws + 18 * MB);
  __bf16* Wvb = (__bf16*)(ws + 20 * MB);
  __bf16* Wob = (__bf16*)(ws + 22 * MB);
  __bf16* Qb  = (__bf16*)(ws + 24 * MB);   // 16 MB
  __bf16* Kb  = (__bf16*)(ws + 40 * MB);   // 16 MB
  __bf16* Vt  = (__bf16*)(ws + 56 * MB);   // 16 MB [B,H,D,T]
  __bf16* Ob  = (__bf16*)(ws + 72 * MB);   // 16 MB attention out

  const int nx = 8192 * 1024;
  const int nw = 1024 * 1024;
  cast_f32_bf16<<<nx / 256, 256, 0, stream>>>(x,  Xb,  nx);
  cast_f32_bf16<<<nw / 256, 256, 0, stream>>>(Wq, Wqb, nw);
  cast_f32_bf16<<<nw / 256, 256, 0, stream>>>(Wk, Wkb, nw);
  cast_f32_bf16<<<nw / 256, 256, 0, stream>>>(Wv, Wvb, nw);
  cast_f32_bf16<<<nw / 256, 256, 0, stream>>>(Wo, Wob, nw);

  dim3 gg(8192 / 64, 1024 / 256);  // (128, 4)
  gemm_xwt<0><<<gg, 256, 0, stream>>>(Xb, Wqb, bq, (void*)Qb);
  gemm_xwt<0><<<gg, 256, 0, stream>>>(Xb, Wkb, bk, (void*)Kb);
  gemm_xwt<1><<<gg, 256, 0, stream>>>(Xb, Wvb, bv, (void*)Vt);

  attn_fwd<<<8192 / 4, 128, 0, stream>>>(Qb, Kb, Vt, Ob);

  gemm_xwt<2><<<gg, 256, 0, stream>>>(Ob, Wob, bo, d_out);
}
